// MultiHeadAttention_7421703487952
// MI455X (gfx1250) — matrix operand order repaired and checked by reading, not compiled, not run on hardware
//
#include <hip/hip_runtime.h>
#include <hip/hip_bf16.h>
#include <math.h>

// Problem constants
#define BB 32
#define NN 512
#define EE 256
#define HH 16
#define DD 16
#define MM (BB * NN)   // 16384 rows

typedef __bf16 v8bf  __attribute__((ext_vector_type(8)));
typedef __bf16 v16bf __attribute__((ext_vector_type(16)));
typedef float  v8f   __attribute__((ext_vector_type(8)));

__device__ __forceinline__ v16bf catbf(v8bf lo, v8bf hi) {
  return __builtin_shufflevector(lo, hi, 0,1,2,3,4,5,6,7,8,9,10,11,12,13,14,15);
}

__device__ __forceinline__ v8bf cvt8(float4 a, float4 b) {
  v8bf r;
  r[0] = (__bf16)a.x; r[1] = (__bf16)a.y; r[2] = (__bf16)a.z; r[3] = (__bf16)a.w;
  r[4] = (__bf16)b.x; r[5] = (__bf16)b.y; r[6] = (__bf16)b.z; r[7] = (__bf16)b.w;
  return r;
}

// 8 contiguous f32 -> 8 bf16 (two global_load_b128)
__device__ __forceinline__ v8bf ld8f(const float* p) {
  float4 a = *(const float4*)p;
  float4 b = *(const float4*)(p + 4);
  return cvt8(a, b);
}
// 16 contiguous f32 -> 16 bf16
__device__ __forceinline__ v16bf ld16f(const float* p) {
  return catbf(ld8f(p), ld8f(p + 8));
}
// 16 contiguous bf16 (two b128 loads)
__device__ __forceinline__ v16bf ld16b(const __bf16* p) {
  return catbf(*(const v8bf*)p, *(const v8bf*)(p + 8));
}

__device__ __forceinline__ v8f wmma_bf16(v16bf a, v16bf b, v8f c) {
  // v_wmma_f32_16x16x32_bf16  (neg_a, A, neg_b, B, c_mod, C, reuse_a, reuse_b)
  return __builtin_amdgcn_wmma_f32_16x16x32_bf16(false, a, false, b, (short)0, c,
                                                 false, false);
}

// Branch-free tanh on the hardware transcendental path:
//   tanh(x) = sign(x) * (e - 1) / (e + 1),  e = exp(2*min(|x|,15))
// Lowers to v_exp_f32 + v_rcp_f32 + FMAs; no EXEC divergence (unlike tanhf).
__device__ __forceinline__ float fast_tanh(float x) {
  float ax = fminf(fabsf(x), 15.0f);
  float e  = __expf(2.0f * ax);
  float t  = (e - 1.0f) * __builtin_amdgcn_rcpf(e + 1.0f);
  return copysignf(t, x);
}

// ---------------------------------------------------------------------------
// GEMM: Y[m,n] = sum_k X[m,k] * W[n,k]   (M x 256) @ (256 x 256)^T
// One wave computes a 16x64 tile (4 WMMA accumulators, K-loop of 8).
// STORE_MODE 0: bf16, row-major (m,e)            [Q, K projections]
// STORE_MODE 1: bf16, transposed (b,h,d,n)       [V projection]
// STORE_MODE 2: f32, row-major (m,e)             [final output]
// IN_BF16: input X is bf16 (else f32, converted inline)
// ---------------------------------------------------------------------------
template <int STORE_MODE, bool IN_BF16>
__global__ __launch_bounds__(32)
void gemm_bt_kernel(const void* __restrict__ Xv, const float* __restrict__ W,
                    void* __restrict__ Out) {
  const int lane  = threadIdx.x;
  const int lr    = lane & 15;   // row (A) / column (B,C) within tile
  const int hi    = lane >> 4;   // which half of the wave
  const int mtile = blockIdx.x * 16;
  const int ctile = blockIdx.y * 64;

  v8f acc[4];
#pragma unroll
  for (int t = 0; t < 4; ++t)
#pragma unroll
    for (int i = 0; i < 8; ++i) acc[t][i] = 0.0f;

  const int arow = mtile + lr;

#pragma unroll
  for (int kt = 0; kt < EE / 32; ++kt) {
    // A fragment: lane-half holds 16 contiguous K values {k0, k0+8}
    const int k0 = kt * 32 + hi * 16;
    const int k1 = k0 + 8;
    v16bf afrag;
    if (IN_BF16) {
      const __bf16* X = (const __bf16*)Xv;
      afrag = catbf(*(const v8bf*)(X + (size_t)arow * EE + k0),
                    *(const v8bf*)(X + (size_t)arow * EE + k1));
    } else {
      const float* X = (const float*)Xv;
      afrag = catbf(ld8f(X + (size_t)arow * EE + k0),
                    ld8f(X + (size_t)arow * EE + k1));
      if (kt + 1 < EE / 32)
        __builtin_prefetch(X + (size_t)arow * EE + k0 + 32, 0, 0);
    }
    // B fragment: lane-half holds 16 contiguous K values for its column
    const int kb = kt * 32 + hi * 16;
#pragma unroll
    for (int t = 0; t < 4; ++t) {
      const int ncol = ctile + t * 16 + lr;
      v16bf bfrag = ld16f(W + (size_t)ncol * EE + kb);
      acc[t] = wmma_bf16(afrag, bfrag, acc[t]);
    }
  }

  // C layout: lane -> column, VGPR i -> row (i + hi*8)
  const int m0 = mtile + hi * 8;
#pragma unroll
  for (int t = 0; t < 4; ++t) {
    const int ncol = ctile + t * 16 + lr;
    if (STORE_MODE == 0) {
      __bf16* O = (__bf16*)Out;
#pragma unroll
      for (int i = 0; i < 8; ++i)
        O[(size_t)(m0 + i) * EE + ncol] = (__bf16)acc[t][i];
    } else if (STORE_MODE == 1) {
      // transposed V: Vt[((b*H + h)*D + d)*N + n], 8 consecutive n -> one b128
      __bf16* O = (__bf16*)Out;
      const int hh = ncol >> 4, dd = ncol & 15;
      const int bb = m0 / NN, n0 = m0 % NN;
      v8bf pk;
#pragma unroll
      for (int i = 0; i < 8; ++i) pk[i] = (__bf16)acc[t][i];
      *(v8bf*)(O + ((size_t)((bb * HH + hh) * DD + dd)) * NN + n0) = pk;
    } else {
      float* O = (float*)Out;
#pragma unroll
      for (int i = 0; i < 8; ++i)
        O[(size_t)(m0 + i) * EE + ncol] = acc[t][i];
    }
  }
}

// ---------------------------------------------------------------------------
// Attention core: one wave handles 16 query rows of one (b, h).
//   logits = 10*tanh(QK^T / 4), key-mask -> -1e10, softmax, @V
// D=16 is zero-padded to K=32 for the bf16 WMMA.
// ---------------------------------------------------------------------------
#define SLOG_STRIDE 516   // float stride, padded to avoid bank conflicts
#define SPRB_STRIDE 520   // bf16 stride, keeps 16B alignment, rotates banks

__global__ __launch_bounds__(32)
void attention_kernel(const __bf16* __restrict__ Qb, const __bf16* __restrict__ Kb,
                      const __bf16* __restrict__ Vt,
                      const unsigned char* __restrict__ mask,
                      __bf16* __restrict__ AO) {
  __shared__ float  slog[16 * SLOG_STRIDE];
  __shared__ __bf16 sprob[16 * SPRB_STRIDE];

  const int lane  = threadIdx.x;
  const int lr    = lane & 15;
  const int hi    = lane >> 4;
  const int bh    = blockIdx.x;
  const int b     = bh >> 4;
  const int h     = bh & 15;
  const int qbase = blockIdx.y * 16;

  v8bf zero8;
#pragma unroll
  for (int i = 0; i < 8; ++i) zero8[i] = (__bf16)0.0f;
  const v16bf zfrag = catbf(zero8, zero8);

  // ---- Phase 1: logits -> LDS ----
  // A fragment = Q rows (d 0..15 split across lane halves, upper K zero-padded)
  const int qrow = qbase + lr;
  v8bf qlo = *(const v8bf*)(Qb + ((size_t)(b * NN + qrow)) * EE + h * DD + hi * 8);
  const v16bf afrag = catbf(qlo, zero8);

#pragma unroll 4
  for (int ct = 0; ct < NN / 16; ++ct) {
    const int kcol = ct * 16 + lr;
    // B fragment: lane-half carries d=hi*8..hi*8+7 for its key column, upper 8 pad zero
    v8bf klo = *(const v8bf*)(Kb + ((size_t)(b * NN + kcol)) * EE + h * DD + hi * 8);
    v16bf bfrag = catbf(klo, zero8);
    v8f c;
#pragma unroll
    for (int i = 0; i < 8; ++i) c[i] = 0.0f;
    c = wmma_bf16(afrag, bfrag, c);

    const bool msk = mask[b * NN + kcol] != 0;
#pragma unroll
    for (int i = 0; i < 8; ++i) {
      float x = 10.0f * fast_tanh(c[i] * 0.25f);  // CLIP*tanh(logit/sqrt(D))
      if (msk) x = -1.0e10f;
      slog[(i + hi * 8) * SLOG_STRIDE + kcol] = x;
    }
  }
  __syncthreads();

  // ---- Phase 2: row softmax (wave-cooperative), repack as bf16 probs ----
  for (int r = 0; r < 16; ++r) {
    float vals[16];
    float mx = -3.0e38f;
#pragma unroll
    for (int j = 0; j < 16; ++j) {
      vals[j] = slog[r * SLOG_STRIDE + lane + j * 32];
      mx = fmaxf(mx, vals[j]);
    }
#pragma unroll
    for (int s = 16; s >= 1; s >>= 1) mx = fmaxf(mx, __shfl_xor(mx, s, 32));
    float sum = 0.0f;
#pragma unroll
    for (int j = 0; j < 16; ++j) { vals[j] = __expf(vals[j] - mx); sum += vals[j]; }
#pragma unroll
    for (int s = 16; s >= 1; s >>= 1) sum += __shfl_xor(sum, s, 32);
    const float inv = __builtin_amdgcn_rcpf(sum);
#pragma unroll
    for (int j = 0; j < 16; ++j)
      sprob[r * SPRB_STRIDE + lane + j * 32] = (__bf16)(vals[j] * inv);
  }
  __syncthreads();

  // ---- Phase 3: out = probs @ V (16 WMMAs over 512 keys) ----
  v8f acc;
#pragma unroll
  for (int i = 0; i < 8; ++i) acc[i] = 0.0f;

  const __bf16* vrow = Vt + ((size_t)((b * HH + h) * DD + lr)) * NN;
#pragma unroll 4
  for (int kt = 0; kt < NN / 32; ++kt) {
    // A = probs fragment from LDS (ds_load_b128 x2)
    const int k0 = kt * 32 + hi * 16;
    v16bf ap = catbf(*(const v8bf*)(sprob + lr * SPRB_STRIDE + k0),
                     *(const v8bf*)(sprob + lr * SPRB_STRIDE + k0 + 8));
    // B = V^T fragment: 16 contiguous keys at head dim d = lr
    v16bf bp = ld16b(vrow + kt * 32 + hi * 16);
    acc = wmma_bf16(ap, bp, acc);
  }

  // store bf16 attention output, combined-heads layout (b, n, h*16+d)
#pragma unroll
  for (int i = 0; i < 8; ++i)
    AO[((size_t)(b * NN + qbase + hi * 8 + i)) * EE + h * DD + lr] = (__bf16)acc[i];
}

// ---------------------------------------------------------------------------
extern "C" void kernel_launch(void* const* d_in, const int* in_sizes, int n_in,
                              void* d_out, int out_size, void* d_ws, size_t ws_size,
                              hipStream_t stream) {
  const float* q    = (const float*)d_in[0];
  const float* k    = (const float*)d_in[1];
  const float* v    = (const float*)d_in[2];
  const unsigned char* mask = (const unsigned char*)d_in[3];
  const float* Wq   = (const float*)d_in[4];
  const float* Wk   = (const float*)d_in[5];
  const float* Wv   = (const float*)d_in[6];
  const float* Wout = (const float*)d_in[7];

  const size_t MN = (size_t)MM * EE;        // 4M elements
  __bf16* Qbf = (__bf16*)d_ws;              // 8 MB
  __bf16* Kbf = Qbf + MN;                   // 8 MB
  __bf16* Vtp = Kbf + MN;                   // 8 MB (transposed V)
  __bf16* AO  = Vtp + MN;                   // 8 MB

  dim3 gg(MM / 16, EE / 64);                // 1024 x 4, 1 wave / block
  gemm_bt_kernel<0, false><<<gg, 32, 0, stream>>>(q, Wq, Qbf);
  gemm_bt_kernel<0, false><<<gg, 32, 0, stream>>>(k, Wk, Kbf);
  gemm_bt_kernel<1, false><<<gg, 32, 0, stream>>>(v, Wv, Vtp);

  attention_kernel<<<dim3(BB * HH, NN / 16), 32, 0, stream>>>(Qbf, Kbf, Vtp, mask, AO);

  gemm_bt_kernel<2, true><<<gg, 32, 0, stream>>>(AO, Wout, d_out);
}
